// MoE_6768868459219
// MI455X (gfx1250) — matrix operand order repaired and checked by reading, compiled, not run on hardware
//
#include <hip/hip_runtime.h>
#include <hip/hip_bf16.h>
#include <math.h>

// ---------------------------------------------------------------------------
// MoE (router + grouped SwiGLU experts + combine) for gfx1250 (MI455X).
// bf16 WMMA (v_wmma_f32_16x16x32_bf16) for all three GEMMs.
// Activation tile staged once in LDS (bf16); h tile fused in LDS.
// ---------------------------------------------------------------------------

typedef __bf16 bf16;
typedef __attribute__((ext_vector_type(16))) __bf16 v16bf;
typedef __attribute__((ext_vector_type(8)))  __bf16 v8bf;
typedef __attribute__((ext_vector_type(8)))  float  v8f;

constexpr int NE   = 8;      // experts
constexpr int TOPK = 2;
constexpr int DIMC = 1024;   // model dim
constexpr int DFFC = 256;    // ffn dim
constexpr int NT   = 32768;  // tokens
constexpr int BM   = 32;     // row tile per workgroup

// ---------------------------------------------------------------------------
// Fragment loaders (CDNA5 16-bit WMMA VGPR layouts, ISA 7.12.2)
// A 16x32: lanes 0-15 -> M=lane, K runs {k0..k0+7, k0+16..k0+23};
//          lanes 16-31 -> M=lane-16, K runs {k0+8..k0+15, k0+24..k0+31}.
// B 32x16: lanes 0-15 -> N=lane, K runs {k0..k0+7, k0+16..k0+23};
//          lanes 16-31 -> N=lane-16, K runs {k0+8..k0+15, k0+24..k0+31}.
// ---------------------------------------------------------------------------

__device__ inline v16bf load_bfrag(const bf16* __restrict__ p) {
  // two runs of 8 bf16 at p and p+16 -> two b128 loads
  const v8bf* q = (const v8bf*)p;
  v8bf lo = q[0], hi = q[2];
  v16bf r;
#pragma unroll
  for (int i = 0; i < 8; ++i) { r[i] = lo[i]; r[i + 8] = hi[i]; }
  return r;
}

__device__ inline v16bf load_afrag_lds(const bf16* __restrict__ row, int runA) {
  const v8bf* p0 = (const v8bf*)(row + runA);
  const v8bf* p1 = (const v8bf*)(row + runA + 16);
  v8bf lo = *p0, hi = *p1;
  v16bf r;
#pragma unroll
  for (int i = 0; i < 8; ++i) { r[i] = lo[i]; r[i + 8] = hi[i]; }
  return r;
}

__device__ inline v8bf cvt8_bf16(float4 a, float4 b) {
  v8bf o;
  o[0] = (bf16)a.x; o[1] = (bf16)a.y; o[2] = (bf16)a.z; o[3] = (bf16)a.w;
  o[4] = (bf16)b.x; o[5] = (bf16)b.y; o[6] = (bf16)b.z; o[7] = (bf16)b.w;
  return o;
}

// ---------------------------------------------------------------------------
// 0) weight prep: fp32 -> bf16 with transpose so B fragments are contiguous.
//    w1,w3: [E][DIM][DFF] -> w1t,w3t: [E][DFF][DIM]
//    w2:    [E][DFF][DIM] -> w2t:     [E][DIM][DFF]
// ---------------------------------------------------------------------------
__global__ __launch_bounds__(256) void moe_prep_w(
    const float* __restrict__ w1, const float* __restrict__ w2,
    const float* __restrict__ w3,
    bf16* __restrict__ w1t, bf16* __restrict__ w2t, bf16* __restrict__ w3t) {
  int i = blockIdx.x * blockDim.x + threadIdx.x;
  if (i >= NE * DFFC * DIMC) return;
  int e   = i / (DFFC * DIMC);
  int rem = i - e * (DFFC * DIMC);
  // w1t/w3t index: [e][n][k]
  int n = rem / DIMC;
  int k = rem - n * DIMC;
  w1t[i] = (bf16)w1[(size_t)e * DIMC * DFFC + (size_t)k * DFFC + n];
  w3t[i] = (bf16)w3[(size_t)e * DIMC * DFFC + (size_t)k * DFFC + n];
  // w2t index: [e][d][f]
  int d = rem / DFFC;
  int f = rem - d * DFFC;
  w2t[i] = (bf16)w2[(size_t)e * DFFC * DIMC + (size_t)f * DIMC + d];
}

// ---------------------------------------------------------------------------
// 1) router: one wave32 per token. logits -> top-2 -> renormalized weights.
// ---------------------------------------------------------------------------
__global__ __launch_bounds__(256) void moe_router(
    const float* __restrict__ x, const float* __restrict__ rw,
    int* __restrict__ sel, float* __restrict__ wts, int* __restrict__ counts) {
  int wave = threadIdx.x >> 5;
  int lane = threadIdx.x & 31;
  int t = blockIdx.x * 8 + wave;
  if (t >= NT) return;

  const float4* xv = (const float4*)(x + (size_t)t * DIMC) + lane * 8;
  const float4* rv = (const float4*)rw;
  float acc[NE];
#pragma unroll
  for (int e = 0; e < NE; ++e) acc[e] = 0.f;
#pragma unroll
  for (int c = 0; c < 8; ++c) {
    float4 xc = xv[c];
#pragma unroll
    for (int e = 0; e < NE; ++e) {
      float4 rc = rv[e * (DIMC / 4) + lane * 8 + c];
      acc[e] += xc.x * rc.x + xc.y * rc.y + xc.z * rc.z + xc.w * rc.w;
    }
  }
#pragma unroll
  for (int e = 0; e < NE; ++e) {
#pragma unroll
    for (int off = 16; off >= 1; off >>= 1)
      acc[e] += __shfl_xor(acc[e], off, 32);
  }
  if (lane == 0) {
    int e0 = 0; float l0 = acc[0];
#pragma unroll
    for (int e = 1; e < NE; ++e) if (acc[e] > l0) { l0 = acc[e]; e0 = e; }
    int e1 = (e0 == 0) ? 1 : 0; float l1 = acc[e1];
#pragma unroll
    for (int e = 0; e < NE; ++e)
      if (e != e0 && acc[e] > l1) { l1 = acc[e]; e1 = e; }
    // softmax over all 8 then renormalize over top-2 == softmax over {l0,l1}
    float w0 = 1.f / (1.f + __expf(l1 - l0));
    sel[t * 2 + 0] = e0; sel[t * 2 + 1] = e1;
    wts[t * 2 + 0] = w0; wts[t * 2 + 1] = 1.f - w0;
    atomicAdd(&counts[e0], 1);
    atomicAdd(&counts[e1], 1);
  }
}

// ---------------------------------------------------------------------------
// 2) exclusive prefix sum of per-expert counts
// ---------------------------------------------------------------------------
__global__ void moe_prefix(const int* __restrict__ counts, int* __restrict__ offsets) {
  if (threadIdx.x == 0 && blockIdx.x == 0) {
    int s = 0;
    for (int e = 0; e < NE; ++e) { offsets[e] = s; s += counts[e]; }
    offsets[NE] = s;
  }
}

// ---------------------------------------------------------------------------
// 3) counting-sort scatter: expert-ordered (token, weight) rows
// ---------------------------------------------------------------------------
__global__ __launch_bounds__(256) void moe_scatter(
    const int* __restrict__ sel, const float* __restrict__ wts,
    const int* __restrict__ offsets, int* __restrict__ cursor,
    int* __restrict__ row_token, float* __restrict__ row_w) {
  int t = blockIdx.x * blockDim.x + threadIdx.x;
  if (t >= NT) return;
#pragma unroll
  for (int k = 0; k < TOPK; ++k) {
    int e = sel[t * 2 + k];
    int pos = offsets[e] + atomicAdd(&cursor[e], 1);
    row_token[pos] = t;
    row_w[pos] = wts[t * 2 + k];
  }
}

// ---------------------------------------------------------------------------
// 4) fused grouped expert: stage x tile in LDS (bf16, once) -> up-proj (w1,w3)
//    -> SwiGLU (h in LDS) -> down-proj (w2) -> weighted atomic combine.
//    8 waves, 32-row tile, full N per workgroup. LDS: 64K (x) + 16K (h).
// ---------------------------------------------------------------------------
__global__ __launch_bounds__(256, 1) void moe_expert_tile(
    const float* __restrict__ x,
    const bf16* __restrict__ w1t, const bf16* __restrict__ w3t,
    const bf16* __restrict__ w2t,
    const int* __restrict__ offsets,
    const int* __restrict__ row_token, const float* __restrict__ row_w,
    float* __restrict__ out) {
  __shared__ __align__(16) bf16 x_lds[BM * DIMC];   // 64 KB
  __shared__ __align__(16) bf16 h_lds[BM * DFFC];   // 16 KB
  __shared__ int   tok_lds[BM];
  __shared__ float wgt_lds[BM];

  const int e    = blockIdx.y;
  const int seg  = offsets[e];
  const int cnt  = offsets[e + 1] - seg;
  const int base = blockIdx.x * BM;
  if (base >= cnt) return;

  const int tid = threadIdx.x;
  if (tid < BM) {
    if (base + tid < cnt) {
      tok_lds[tid] = row_token[seg + base + tid];
      wgt_lds[tid] = row_w[seg + base + tid];
    } else {
      tok_lds[tid] = -1;
      wgt_lds[tid] = 0.f;
    }
  }
  __syncthreads();

  // -------- stage activation tile: 32 rows x 1024, fp32 -> bf16, once ------
  {
    const int r  = tid >> 3;            // 8 threads per row
    int tok = tok_lds[r];
    if (tok < 0) tok = 0;               // pad rows: garbage, skipped at epilogue
    const float4* src = (const float4*)(x + (size_t)tok * DIMC);
    bf16* dst = x_lds + r * DIMC;
    const int c4 = (tid & 7) * 32;      // this thread: float4s [c4, c4+32)
#pragma unroll
    for (int u = 0; u < 16; ++u) {
      float4 a = src[c4 + 2 * u];
      float4 b = src[c4 + 2 * u + 1];
      *(v8bf*)(dst + 4 * (c4 + 2 * u)) = cvt8_bf16(a, b);
    }
  }
  __syncthreads();

  const int wave = tid >> 5;
  const int lane = tid & 31;
  const int mt   = wave & 1;    // which 16-row half of the 32-row tile
  const int pp   = wave >> 1;   // wave pair id, 0..3
  const int nl   = lane & 15;
  const int hiL  = lane >> 4;   // 0 or 1
  const int m_local = mt * 16 + nl;       // A row this lane loads
  const int koffB   = hiL ? 8 : 0;        // B fragment K offset
  const int runAoff = hiL ? 8 : 0;        // A fragment first-run K offset

  const bf16* w1e = w1t + (size_t)e * DFFC * DIMC;
  const bf16* w3e = w3t + (size_t)e * DFFC * DIMC;
  const bf16* w2e = w2t + (size_t)e * DIMC * DFFC;

  // ---------------- up projection: [32 x 1024] x [1024 x 256] (w1 and w3) ---
  v8f acc1[4], acc3[4];
#pragma unroll
  for (int j = 0; j < 4; ++j) {
    acc1[j] = (v8f)(0.f);
    acc3[j] = (v8f)(0.f);
  }

  for (int k0 = 0; k0 < DIMC; k0 += 32) {
    v16bf a = load_afrag_lds(x_lds + m_local * DIMC, k0 + runAoff);
    __builtin_prefetch(w1e + (size_t)(4 * pp * 16 + nl) * DIMC + k0 + 64, 0, 0);
#pragma unroll
    for (int j = 0; j < 4; ++j) {
      const int n = (4 * pp + j) * 16 + nl;
      v16bf b1 = load_bfrag(w1e + (size_t)n * DIMC + k0 + koffB);
      acc1[j] = __builtin_amdgcn_wmma_f32_16x16x32_bf16(
          false, a, false, b1, (short)0, acc1[j], false, false);
      v16bf b3 = load_bfrag(w3e + (size_t)n * DIMC + k0 + koffB);
      acc3[j] = __builtin_amdgcn_wmma_f32_16x16x32_bf16(
          false, a, false, b3, (short)0, acc3[j], false, false);
    }
  }

  // -------- SwiGLU in f32 (fast rcp), write h tile (bf16) to LDS ------------
  // D layout: lane -> N=nl, vgpr i -> M = mt*16 + hiL*8 + i
#pragma unroll
  for (int j = 0; j < 4; ++j) {
    const int n = (4 * pp + j) * 16 + nl;
#pragma unroll
    for (int i = 0; i < 8; ++i) {
      float g = acc1[j][i];
      float s = __builtin_amdgcn_rcpf(1.f + __expf(-g));  // sigmoid(g)
      float h = g * s * acc3[j][i];                       // silu(x1)*x3
      const int m = mt * 16 + hiL * 8 + i;
      h_lds[m * DFFC + n] = (bf16)h;
    }
  }
  __syncthreads();

  // ---------------- down projection: [32 x 256] x [256 x 1024] --------------
  v8f acc[16];
#pragma unroll
  for (int j = 0; j < 16; ++j) acc[j] = (v8f)(0.f);

  for (int k0 = 0; k0 < DFFC; k0 += 32) {
    v16bf a = load_afrag_lds(h_lds + m_local * DFFC, k0 + runAoff);
#pragma unroll
    for (int j = 0; j < 16; ++j) {
      const int n = (16 * pp + j) * 16 + nl;
      v16bf b = load_bfrag(w2e + (size_t)n * DFFC + k0 + koffB);
      acc[j] = __builtin_amdgcn_wmma_f32_16x16x32_bf16(
          false, a, false, b, (short)0, acc[j], false, false);
    }
  }

  // -------- combine: weighted atomic scatter-add into out -------------------
#pragma unroll
  for (int j = 0; j < 16; ++j) {
    const int n = (16 * pp + j) * 16 + nl;
#pragma unroll
    for (int i = 0; i < 8; ++i) {
      const int m = mt * 16 + hiL * 8 + i;
      const int tok = tok_lds[m];
      if (tok >= 0)
        atomicAdd(&out[(size_t)tok * DIMC + n], acc[j][i] * wgt_lds[m]);
    }
  }
}

// ---------------------------------------------------------------------------
// host launcher
// ---------------------------------------------------------------------------
extern "C" void kernel_launch(void* const* d_in, const int* in_sizes, int n_in,
                              void* d_out, int out_size, void* d_ws, size_t ws_size,
                              hipStream_t stream) {
  const float* x  = (const float*)d_in[0];
  const float* rw = (const float*)d_in[1];
  const float* w1 = (const float*)d_in[2];
  const float* w2 = (const float*)d_in[3];
  const float* w3 = (const float*)d_in[4];
  float* out = (float*)d_out;

  char* ws = (char*)d_ws;
  size_t off = 0;
  auto walloc = [&](size_t bytes) -> char* {
    char* p = ws + off;
    off += (bytes + 255) & ~size_t(255);
    return p;
  };
  bf16*  w1t       = (bf16*)walloc(sizeof(bf16) * NE * DFFC * DIMC);
  bf16*  w3t       = (bf16*)walloc(sizeof(bf16) * NE * DFFC * DIMC);
  bf16*  w2t       = (bf16*)walloc(sizeof(bf16) * NE * DIMC * DFFC);
  int*   sel       = (int*)walloc(sizeof(int) * NT * TOPK);
  float* wts       = (float*)walloc(sizeof(float) * NT * TOPK);
  int*   row_token = (int*)walloc(sizeof(int) * NT * TOPK);
  float* row_w     = (float*)walloc(sizeof(float) * NT * TOPK);
  int*   counts    = (int*)walloc(sizeof(int) * 64);
  int*   cursor    = counts + 8;
  int*   offsets   = counts + 16;  // [NE+1]

  hipMemsetAsync(counts, 0, sizeof(int) * 64, stream);
  hipMemsetAsync(out, 0, (size_t)out_size * sizeof(float), stream);

  moe_prep_w<<<(NE * DFFC * DIMC + 255) / 256, 256, 0, stream>>>(
      w1, w2, w3, w1t, w2t, w3t);
  moe_router<<<NT / 8, 256, 0, stream>>>(x, rw, sel, wts, counts);
  moe_prefix<<<1, 32, 0, stream>>>(counts, offsets);
  moe_scatter<<<NT / 256, 256, 0, stream>>>(sel, wts, offsets, cursor,
                                            row_token, row_w);
  dim3 grid(NT / BM, NE);
  moe_expert_tile<<<grid, 256, 0, stream>>>(x, w1t, w3t, w2t, offsets,
                                            row_token, row_w, out);
}
